// Decoder_2886218023676
// MI455X (gfx1250) — matrix operand order repaired and checked by reading, not compiled, not run on hardware
//
#include <hip/hip_runtime.h>
#include <hip/hip_bf16.h>

// ---------------------------------------------------------------------------
// GRU decoder rollout for MI455X (gfx1250, wave32, WMMA).
//   B=64, FC=24, N=1000, EMB=32, HID=64, 3H=192, IN=33 (x_prev + emb)
// One wave32 owns a 16-row tile for all 24 timesteps.
//  - GEMMs via v_wmma_f32_16x16x32_f16 (36 per step per wave)
//  - x_prev column handled as a scalar rank-1 update fused into the bias init
//  - hidden state carried in f32 registers (C/D layout); f16 shadow in LDS
//    only for the WMMA A-fragments
//  - small constant tables (biases / x_prev column / W_out) stay in LDS to
//    keep VGPR pressure (and s_set_vgpr_msb churn) down
//  - gate nonlinearities on HW transcendentals (v_exp_f32/v_rcp_f32/v_tanh_f32)
// ---------------------------------------------------------------------------

typedef __attribute__((ext_vector_type(16))) _Float16 v16h;
typedef __attribute__((ext_vector_type(8)))  _Float16 v8h;
typedef __attribute__((ext_vector_type(8)))  float    v8f;
typedef __attribute__((ext_vector_type(4)))  float    v4f;

#define B_    64
#define HIST  24
#define FC_   24
#define N_    1000
#define F_    8
#define EMB_  32
#define HID_  64
#define G3    192                 // 3*HID
#define TOTROWS (B_ * N_)         // 64000
#define WAVES 8
#define ROWS  16
#define BLOCK (WAVES * 32)        // 256 threads
#define ROWS_PER_BLOCK (WAVES * ROWS)          // 128
#define NBLOCKS (TOTROWS / ROWS_PER_BLOCK)     // 500

#define LOG2E_F 1.44269504088896340736f

union AFrag { v16h v; v8h h[2]; };

static __device__ __forceinline__ v8f wmma_f16(v16h a, v16h b, v8f c) {
    // D = A(16x32 f16) * B(32x16 f16) + C(16x16 f32)
    return __builtin_amdgcn_wmma_f32_16x16x32_f16(
        /*neg_a=*/false, a, /*neg_b=*/false, b,
        /*c_mod=*/(short)0, c, /*reuse_a=*/false, /*reuse_b=*/false);
}

// sigmoid(x) = rcp(1 + exp2(-x*log2(e)))  -> v_exp_f32 + v_rcp_f32
static __device__ __forceinline__ float fast_sigmoid(float x) {
    const float e = __builtin_amdgcn_exp2f(-x * LOG2E_F);
    return __builtin_amdgcn_rcpf(1.0f + e);
}

#if __has_builtin(__builtin_amdgcn_tanhf)
static __device__ __forceinline__ float fast_tanh(float x) {
    return __builtin_amdgcn_tanhf(x);          // native v_tanh_f32 on gfx1250
}
#else
static __device__ __forceinline__ float fast_tanh(float x) {
    // tanh(x) = 2*sigmoid(2x) - 1
    const float e = __builtin_amdgcn_exp2f(x * (-2.0f * LOG2E_F));
    return 2.0f * __builtin_amdgcn_rcpf(1.0f + e) - 1.0f;
}
#endif

__global__ __launch_bounds__(BLOCK)
void gru_decoder_kernel(const float* __restrict__ X,
                        const float* __restrict__ hn,
                        const float* __restrict__ xn,
                        const float* __restrict__ emb_table,
                        const float* __restrict__ Wih,   // [192][33]
                        const float* __restrict__ Whh,   // [192][64]
                        const float* __restrict__ bih,   // [192]
                        const float* __restrict__ bhh,   // [192]
                        const float* __restrict__ Wout,  // [1][64]
                        const float* __restrict__ boutp, // [1]
                        float* __restrict__ out)         // [B][FC][N]
{
    // ---- LDS (62.5 KB) -----------------------------------------------------
    __shared__ __align__(16) _Float16 sWih[G3][EMB_];   // col k == Wih[n][1+k]
    __shared__ __align__(16) _Float16 sWhh[G3][HID_];
    __shared__ __align__(16) float    sWih0[G3];        // Wih[n][0] (x_prev col)
    __shared__ __align__(16) float    sBrz[2 * HID_];   // bih+bhh for r,z
    __shared__ __align__(16) float    sBin[HID_];       // bih n-gate
    __shared__ __align__(16) float    sBhn[HID_];       // bhh n-gate
    __shared__ __align__(16) float    sWout[HID_];
    __shared__ __align__(16) _Float16 sX[WAVES][ROWS][EMB_];  // emb tile (f16)
    __shared__ __align__(16) _Float16 sH[WAVES][ROWS][HID_];  // h shadow (f16)
    __shared__ __align__(16) float    sXp[WAVES][ROWS];       // x_prev (f32)

    const int tid  = threadIdx.x;
    const int wave = tid >> 5;
    const int lane = tid & 31;

    // ---- Stage weights + constants once per block (f32 -> f16) ------------
    for (int i = tid; i < G3 * EMB_; i += BLOCK) {
        const int n = i >> 5, k = i & 31;
        sWih[n][k] = (_Float16)Wih[n * 33 + 1 + k];
    }
    for (int i = tid; i < G3; i += BLOCK) sWih0[i] = Wih[i * 33];
    for (int i = tid; i < G3 * HID_; i += BLOCK) {
        const int n = i >> 6, k = i & 63;
        sWhh[n][k] = (_Float16)Whh[n * HID_ + k];
    }
    for (int i = tid; i < 2 * HID_; i += BLOCK) sBrz[i] = bih[i] + bhh[i];
    for (int i = tid; i < HID_; i += BLOCK) {
        sBin[i]  = bih[2 * HID_ + i];
        sBhn[i]  = bhh[2 * HID_ + i];
        sWout[i] = Wout[i];
    }

    // ---- Per-wave tile init ------------------------------------------------
    const int row = lane & 15;           // row within 16-row tile
    const int cg  = lane >> 4;           // column-group 0/1 for staging
    const int tile_base = blockIdx.x * ROWS_PER_BLOCK + wave * ROWS;
    const int rid = tile_base + row;     // this lane's carried row id

    {   // h0: 32 floats per lane -> f16 LDS shadow
        const float* hp = hn + (size_t)rid * HID_ + cg * 32;
        #pragma unroll
        for (int j = 0; j < 32; ++j)
            sH[wave][row][cg * 32 + j] = (_Float16)hp[j];
    }
    if (cg == 0) sXp[wave][row] = xn[rid];
    __syncthreads();

    const float bout = boutp[0];

    // ---- Lane geometry for WMMA fragments ---------------------------------
    const int Mbase = (lane < 16) ? 0 : 8;   // C/D rows this lane holds
    const int kA    = (lane < 16) ? 0 : 16;  // A-frag K base
    const int kB    = (lane < 16) ? 0 : 16;  // B-frag K base
    const int col   = lane & 15;             // C/D column within 16-wide tile

    // ---- f32 hidden state in registers, C/D layout: [t][r] -> (Mbase+r, t*16+col)
    float hstate[4][8];
    #pragma unroll
    for (int t = 0; t < 4; ++t)
        #pragma unroll
        for (int r = 0; r < 8; ++r)
            hstate[t][r] = hn[(size_t)(tile_base + Mbase + r) * HID_ + t * 16 + col];

    const int bb = rid / N_;
    const int nn = rid - bb * N_;
    const size_t idx_base = ((size_t)bb * (HIST + FC_)) * N_ * F_ + (size_t)nn * F_ + (F_ - 1);

    for (int step = 0; step < FC_; ++step) {
        // ---- embedding gather -> sX (each lane: 16 floats -> 16 f16) ------
        {
            const int idx = (int)X[idx_base + (size_t)step * N_ * F_];
            const v4f* ep = (const v4f*)(emb_table + (size_t)idx * EMB_ + cg * 16);
            const v4f e0 = ep[0], e1 = ep[1], e2 = ep[2], e3 = ep[3];
            v8h p0, p1;
            #pragma unroll
            for (int j = 0; j < 4; ++j) {
                p0[j]     = (_Float16)e0[j];
                p0[4 + j] = (_Float16)e1[j];
                p1[j]     = (_Float16)e2[j];
                p1[4 + j] = (_Float16)e3[j];
            }
            *(v8h*)&sX[wave][row][cg * 16]     = p0;
            *(v8h*)&sX[wave][row][cg * 16 + 8] = p1;
        }
        __builtin_amdgcn_wave_barrier();   // order LDS produce vs consume

        // ---- A fragments (contiguous K order, same as B: lanes<16 K{0-15}) ---------
        AFrag ax, ah0, ah1;
        ax.h[0]  = *(const v8h*)&sX[wave][row][kA];
        ax.h[1]  = *(const v8h*)&sX[wave][row][kA + 8];
        ah0.h[0] = *(const v8h*)&sH[wave][row][kA];
        ah0.h[1] = *(const v8h*)&sH[wave][row][kA + 8];
        ah1.h[0] = *(const v8h*)&sH[wave][row][32 + kA];
        ah1.h[1] = *(const v8h*)&sH[wave][row][40 + kA];

        float xp[8];
        #pragma unroll
        for (int r = 0; r < 8; ++r) xp[r] = sXp[wave][Mbase + r];

        float outp[8] = {0.f, 0.f, 0.f, 0.f, 0.f, 0.f, 0.f, 0.f};

        // ---- 4 gate-column slices of 16 -----------------------------------
        #pragma unroll
        for (int t = 0; t < 4; ++t) {
            const int c = t * 16 + col;          // hidden column this lane owns
            const float br = sBrz[c],        bz = sBrz[HID_ + c];
            const float bi = sBin[c],        bh = sBhn[c];
            const float w0r = sWih0[c], w0z = sWih0[HID_ + c], w0n = sWih0[2 * HID_ + c];
            const float wo  = sWout[c];

            v8f accr, accz, accin, acchn;
            #pragma unroll
            for (int r = 0; r < 8; ++r) {        // bias + x_prev rank-1 term
                accr[r]  = br + xp[r] * w0r;
                accz[r]  = bz + xp[r] * w0z;
                accin[r] = bi + xp[r] * w0n;
                acchn[r] = bh;
            }

            AFrag bir, bhr0, bhr1, biz, bhz0, bhz1, bin_, bhn0, bhn1;
            // r-gate weights (rows c)
            bir.h[0]  = *(const v8h*)&sWih[c][kB];
            bir.h[1]  = *(const v8h*)&sWih[c][kB + 8];
            bhr0.h[0] = *(const v8h*)&sWhh[c][kB];
            bhr0.h[1] = *(const v8h*)&sWhh[c][kB + 8];
            bhr1.h[0] = *(const v8h*)&sWhh[c][32 + kB];
            bhr1.h[1] = *(const v8h*)&sWhh[c][32 + kB + 8];
            // z-gate weights (rows 64+c)
            biz.h[0]  = *(const v8h*)&sWih[HID_ + c][kB];
            biz.h[1]  = *(const v8h*)&sWih[HID_ + c][kB + 8];
            bhz0.h[0] = *(const v8h*)&sWhh[HID_ + c][kB];
            bhz0.h[1] = *(const v8h*)&sWhh[HID_ + c][kB + 8];
            bhz1.h[0] = *(const v8h*)&sWhh[HID_ + c][32 + kB];
            bhz1.h[1] = *(const v8h*)&sWhh[HID_ + c][32 + kB + 8];
            // n-gate weights (rows 128+c)
            bin_.h[0] = *(const v8h*)&sWih[2 * HID_ + c][kB];
            bin_.h[1] = *(const v8h*)&sWih[2 * HID_ + c][kB + 8];
            bhn0.h[0] = *(const v8h*)&sWhh[2 * HID_ + c][kB];
            bhn0.h[1] = *(const v8h*)&sWhh[2 * HID_ + c][kB + 8];
            bhn1.h[0] = *(const v8h*)&sWhh[2 * HID_ + c][32 + kB];
            bhn1.h[1] = *(const v8h*)&sWhh[2 * HID_ + c][32 + kB + 8];

            // 9 WMMAs: r(3), z(3), i_n(1), h_n(2)
            accr  = wmma_f16(ax.v,  bir.v,  accr);
            accr  = wmma_f16(ah0.v, bhr0.v, accr);
            accr  = wmma_f16(ah1.v, bhr1.v, accr);
            accz  = wmma_f16(ax.v,  biz.v,  accz);
            accz  = wmma_f16(ah0.v, bhz0.v, accz);
            accz  = wmma_f16(ah1.v, bhz1.v, accz);
            accin = wmma_f16(ax.v,  bin_.v, accin);
            acchn = wmma_f16(ah0.v, bhn0.v, acchn);
            acchn = wmma_f16(ah1.v, bhn1.v, acchn);

            // ---- gate math; h carried in f32 regs, f16 shadow to LDS -------
            #pragma unroll
            for (int r = 0; r < 8; ++r) {
                const float rr = fast_sigmoid(accr[r]);
                const float zz = fast_sigmoid(accz[r]);
                const float nv = fast_tanh(accin[r] + rr * acchn[r]);
                const float hv = (1.0f - zz) * nv + zz * hstate[t][r];
                hstate[t][r] = hv;
                sH[wave][Mbase + r][c] = (_Float16)hv;
                outp[r] += wo * hv;
            }
        }

        // ---- output head: reduce over 16 hidden-column lanes --------------
        #pragma unroll
        for (int r = 0; r < 8; ++r) {
            float v = outp[r];
            v += __shfl_xor(v, 1, 32);
            v += __shfl_xor(v, 2, 32);
            v += __shfl_xor(v, 4, 32);
            v += __shfl_xor(v, 8, 32);
            outp[r] = v + bout;
        }
        if (col == 0) {   // lanes 0 and 16 write rows Mbase..Mbase+7
            #pragma unroll
            for (int r = 0; r < 8; ++r) {
                const int M    = Mbase + r;
                const int grow = tile_base + M;
                const int gb   = grow / N_;
                const int gn   = grow - gb * N_;
                out[((size_t)gb * FC_ + step) * N_ + gn] = outp[r];
                sXp[wave][M] = outp[r];          // becomes next x_prev
            }
        }
        __builtin_amdgcn_wave_barrier();
    }
}

extern "C" void kernel_launch(void* const* d_in, const int* in_sizes, int n_in,
                              void* d_out, int out_size, void* d_ws, size_t ws_size,
                              hipStream_t stream) {
    (void)in_sizes; (void)n_in; (void)out_size; (void)d_ws; (void)ws_size;
    const float* X    = (const float*)d_in[0];
    const float* hn   = (const float*)d_in[1];
    const float* xn   = (const float*)d_in[2];
    const float* emb  = (const float*)d_in[3];
    const float* Wih  = (const float*)d_in[4];
    const float* Whh  = (const float*)d_in[5];
    const float* bih  = (const float*)d_in[6];
    const float* bhh  = (const float*)d_in[7];
    const float* Wout = (const float*)d_in[8];
    const float* bout = (const float*)d_in[9];
    float* out = (float*)d_out;

    hipLaunchKernelGGL(gru_decoder_kernel, dim3(NBLOCKS), dim3(BLOCK), 0, stream,
                       X, hn, xn, emb, Wih, Whh, bih, bhh, Wout, bout, out);
}
